// SpatiotemporalMambaBlock_309237645383
// MI455X (gfx1250) — matrix operand order repaired and checked by reading, not compiled, not run on hardware
//
#include <hip/hip_runtime.h>
#include <hip/hip_bf16.h>
#include <stdint.h>

typedef __attribute__((ext_vector_type(16))) __bf16 v16bf;
typedef __attribute__((ext_vector_type(8)))  float  v8f;

#if __has_builtin(__builtin_amdgcn_global_load_async_to_lds_b128) && \
    __has_builtin(__builtin_amdgcn_s_wait_asynccnt)
#define HAVE_ASYNC_LDS 1
// exact param-1 type from hipcc diagnostic: v4i addrspace(1)*
typedef __attribute__((__vector_size__(4 * sizeof(int)))) int v4i_;
typedef __attribute__((address_space(1))) v4i_* as1_v4i;
typedef __attribute__((address_space(3))) v4i_* as3_v4i;
#else
#define HAVE_ASYNC_LDS 0
#endif

// ---------------- problem constants ----------------
#define B_SZ        4
#define T_SZ        8
#define L_SEQ       8192          // T * 32 * 32
#define NTOK        32768         // B * L
#define DM          256
#define D_INNER     512
#define D_STATE     16
#define NHEADS      8
#define HEADDIM     64
#define CONV_DIM    544
#define D_IN_PROJ   1064
#define KPATCH      368
#define KPATCH_PAD  384
#define NPROJ_PAD   1088
#define NUN_PAD     384

// ---------------- GEMM tiling ----------------
#define GBLK_M 64
#define GBLK_N 64
#define LDSS   40      // padded LDS row stride in bf16 elems (80B, 16B aligned)

// ---------------- workspace layout (bytes, all 256-aligned) ----------------
static constexpr size_t OFF_APATCH = 0;                                            // 32768*384*2
static constexpr size_t OFF_BPATCH = OFF_APATCH + (size_t)NTOK * KPATCH_PAD * 2;   // 384*256*2
static constexpr size_t OFF_XS     = OFF_BPATCH + (size_t)KPATCH_PAD * DM * 2;     // 32768*256*2
static constexpr size_t OFF_WINB   = OFF_XS     + (size_t)NTOK * DM * 2;           // 256*1088*2
static constexpr size_t OFF_ZX     = OFF_WINB   + (size_t)DM * NPROJ_PAD * 2;      // 32768*1064*4
static constexpr size_t OFF_XBC    = OFF_ZX     + (size_t)NTOK * D_IN_PROJ * 4;    // 32768*544*4
static constexpr size_t OFF_DT     = OFF_XBC    + (size_t)NTOK * CONV_DIM * 4;     // 32768*8*4
static constexpr size_t OFF_DA     = OFF_DT     + (size_t)NTOK * NHEADS * 4;
static constexpr size_t OFF_YB     = OFF_DA     + (size_t)NTOK * NHEADS * 4;       // 32768*512*4
static constexpr size_t OFF_YN     = OFF_YB     + (size_t)NTOK * D_INNER * 4;      // 32768*512*2
static constexpr size_t OFF_WOUTB  = OFF_YN     + (size_t)NTOK * D_INNER * 2;      // 512*256*2
static constexpr size_t OFF_OSEQ   = OFF_WOUTB  + (size_t)D_INNER * DM * 2;        // 32768*256*2
static constexpr size_t OFF_BUN    = OFF_OSEQ   + (size_t)NTOK * DM * 2;           // 256*384*2

// =====================================================================
// WMMA bf16 GEMM tile: C[64x64 per block] = A[M,KTOT] * B[KTOT,NPAD]
// 256 threads = 8 waves as 4(M) x 2(N); each wave computes 16x32
// (two v_wmma_f32_16x16x32_bf16 per 32-wide K step).
// =====================================================================
template<int KTOT, int NPAD>
__device__ __forceinline__ void gemm_tile(const __bf16* __restrict__ A,
                                          const __bf16* __restrict__ B,
                                          int blockM, int blockN,
                                          __bf16* lds_a, __bf16* lds_bt,
                                          v8f& acc0, v8f& acc1)
{
    const int tid   = threadIdx.x;
    const int lane  = tid & 31;
    const int wave  = tid >> 5;
    const int waveM = wave >> 1;        // 0..3
    const int waveN = wave & 1;         // 0..1
    const int mh    = lane & 15;
    const int khalf = lane >> 4;

    const int ar  = tid >> 2;           // A tile row 0..63
    const int akc = tid & 3;            // A 8-elem chunk 0..3
    const int bk  = tid >> 3;           // B tile k row 0..31
    const int bn8 = (tid & 7) << 3;     // B 8-col group 0..56

    for (int k0 = 0; k0 < KTOT; k0 += 32) {
        // ---- stage A (row-major): one 16B chunk per thread
#if HAVE_ASYNC_LDS
        __builtin_amdgcn_global_load_async_to_lds_b128(
            (as1_v4i)(A + (size_t)(blockM + ar) * KTOT + k0 + akc * 8),
            (as3_v4i)(lds_a + ar * LDSS + akc * 8),
            0, 0);
#else
        *(uint4*)(lds_a + ar * LDSS + akc * 8) =
            *(const uint4*)(A + (size_t)(blockM + ar) * KTOT + k0 + akc * 8);
#endif
        // ---- stage B transposed: lds_bt[n][k], 8 bf16 per thread
        {
            const __bf16* src = B + (size_t)(k0 + bk) * NPAD + blockN + bn8;
            __bf16 t8[8];
            *(uint4*)t8 = *(const uint4*)src;
#pragma unroll
            for (int i = 0; i < 8; ++i)
                lds_bt[(bn8 + i) * LDSS + bk] = t8[i];
        }
#if HAVE_ASYNC_LDS
        __builtin_amdgcn_s_wait_asynccnt(0);
#endif
        __syncthreads();

        // A fragment (ISA 7.12.2, 16-bit A 16x32): lane m holds
        // K = khalf*8 + [0..7]  and  K = 16 + khalf*8 + [0..7]
        union { v16bf v; uint4 u[2]; } af, bf0, bf1;
        const __bf16* arow = lds_a + (waveM * 16 + mh) * LDSS;
        af.u[0] = *(const uint4*)(arow + khalf * 8);
        af.u[1] = *(const uint4*)(arow + 16 + khalf * 8);
        // B fragment (32x16): lane n holds K = khalf*8 + [0..7] and K = 16 + khalf*8 + [0..7]
        const __bf16* brow0 = lds_bt + (waveN * 32 + mh) * LDSS + khalf * 8;
        const __bf16* brow1 = lds_bt + (waveN * 32 + 16 + mh) * LDSS + khalf * 8;
        bf0.u[0] = *(const uint4*)(brow0);
        bf0.u[1] = *(const uint4*)(brow0 + 16);
        bf1.u[0] = *(const uint4*)(brow1);
        bf1.u[1] = *(const uint4*)(brow1 + 16);

        acc0 = __builtin_amdgcn_wmma_f32_16x16x32_bf16(false, af.v, false, bf0.v,
                                                       (short)0, acc0, false, false);
        acc1 = __builtin_amdgcn_wmma_f32_16x16x32_bf16(false, af.v, false, bf1.v,
                                                       (short)0, acc1, false, false);
        __syncthreads();
    }
}

// C/D layout: col n = lane&15, row m = vgpr r + 8*(lane>>4)
#define GEMM_EPILOG_COORDS()                                        \
    const int lane = threadIdx.x & 31;                              \
    const int wave = threadIdx.x >> 5;                              \
    const int gm0  = blockM + (wave >> 1) * 16 + (lane >> 4) * 8;   \
    const int gn0  = blockN + (wave & 1) * 32 + (lane & 15);        \
    const int gn1  = gn0 + 16;

// ---------------------------------------------------------------------
// 1) im2col: x (B,23,T,128,128) -> bf16 patches [NTOK, 384] (cols >=368 zero)
__global__ void k_im2col(const float* __restrict__ x, __bf16* __restrict__ Ap)
{
    size_t idx = (size_t)blockIdx.x * blockDim.x + threadIdx.x;
    if (idx >= (size_t)NTOK * KPATCH_PAD) return;
    int    kk = (int)(idx % KPATCH_PAD);
    size_t m  = idx / KPATCH_PAD;
    float v = 0.f;
    if (kk < KPATCH) {
        int b = (int)(m >> 13), r = (int)(m & 8191);
        int t = r >> 10, rr = r & 1023, hp = rr >> 5, wp = rr & 31;
        int c = kk >> 4, q = kk & 15, i = q >> 2, j = q & 3;
        size_t src = ((((size_t)(b * 23 + c) * T_SZ + t) * 128 + hp * 4 + i) * 128)
                     + wp * 4 + j;
        v = x[src];
    }
    Ap[idx] = (__bf16)v;
}

// 2) patch_w (DM,23,4,4) -> Bpatch bf16 [384][256]
__global__ void k_prep_bpatch(const float* __restrict__ pw, __bf16* __restrict__ Bp)
{
    int idx = blockIdx.x * blockDim.x + threadIdx.x;
    if (idx >= KPATCH_PAD * DM) return;
    int kk = idx / DM, d = idx % DM;
    float v = 0.f;
    if (kk < KPATCH) {
        int c = kk >> 4, q = kk & 15, i = q >> 2, j = q & 3;
        v = pw[(((size_t)d * 23 + c) * 4 + i) * 4 + j];
    }
    Bp[idx] = (__bf16)v;
}

// 3) W_in (256,1064) -> bf16 [256][1088] zero-padded
__global__ void k_prep_win(const float* __restrict__ w, __bf16* __restrict__ o)
{
    int idx = blockIdx.x * blockDim.x + threadIdx.x;
    if (idx >= DM * NPROJ_PAD) return;
    int k = idx / NPROJ_PAD, n = idx % NPROJ_PAD;
    o[idx] = (__bf16)((n < D_IN_PROJ) ? w[(size_t)k * D_IN_PROJ + n] : 0.f);
}

// 4) W_out (512,256) -> bf16
__global__ void k_prep_wout(const float* __restrict__ w, __bf16* __restrict__ o)
{
    int idx = blockIdx.x * blockDim.x + threadIdx.x;
    if (idx >= D_INNER * DM) return;
    o[idx] = (__bf16)w[idx];
}

// 5) unembed_w (DM,23,4,4) -> Bun bf16 [256][384]
__global__ void k_prep_bun(const float* __restrict__ uw, __bf16* __restrict__ o)
{
    int idx = blockIdx.x * blockDim.x + threadIdx.x;
    if (idx >= DM * NUN_PAD) return;
    int d = idx / NUN_PAD, kk = idx % NUN_PAD;
    float v = 0.f;
    if (kk < KPATCH) {
        int c = kk >> 4, q = kk & 15, i = q >> 2, j = q & 3;
        v = uw[(((size_t)d * 23 + c) * 4 + i) * 4 + j];
    }
    o[idx] = (__bf16)v;
}

// ---------------------------------------------------------------------
// GEMM 1: xs = patches @ Bpatch + patch_b  -> bf16 [NTOK, 256]
__global__ void __launch_bounds__(256)
k_gemm_patch(const __bf16* __restrict__ A, const __bf16* __restrict__ B,
             const float* __restrict__ bias, __bf16* __restrict__ xs)
{
    __shared__ __align__(16) __bf16 sm[(GBLK_M + GBLK_N) * LDSS];
    const int blockM = blockIdx.y * GBLK_M, blockN = blockIdx.x * GBLK_N;
    v8f acc0 = {}, acc1 = {};
    gemm_tile<KPATCH_PAD, DM>(A, B, blockM, blockN, sm, sm + GBLK_M * LDSS, acc0, acc1);
    GEMM_EPILOG_COORDS();
    float bv0 = bias[gn0], bv1 = bias[gn1];
#pragma unroll
    for (int r = 0; r < 8; ++r) {
        xs[(size_t)(gm0 + r) * DM + gn0] = (__bf16)(acc0[r] + bv0);
        xs[(size_t)(gm0 + r) * DM + gn1] = (__bf16)(acc1[r] + bv1);
    }
}

// GEMM 2: zxbcdt = xs @ W_in  -> fp32 [NTOK, 1064]
__global__ void __launch_bounds__(256)
k_gemm_inproj(const __bf16* __restrict__ A, const __bf16* __restrict__ B,
              float* __restrict__ zx)
{
    __shared__ __align__(16) __bf16 sm[(GBLK_M + GBLK_N) * LDSS];
    const int blockM = blockIdx.y * GBLK_M, blockN = blockIdx.x * GBLK_N;
    v8f acc0 = {}, acc1 = {};
    gemm_tile<DM, NPROJ_PAD>(A, B, blockM, blockN, sm, sm + GBLK_M * LDSS, acc0, acc1);
    GEMM_EPILOG_COORDS();
#pragma unroll
    for (int r = 0; r < 8; ++r) {
        if (gn0 < D_IN_PROJ) zx[(size_t)(gm0 + r) * D_IN_PROJ + gn0] = acc0[r];
        if (gn1 < D_IN_PROJ) zx[(size_t)(gm0 + r) * D_IN_PROJ + gn1] = acc1[r];
    }
}

// GEMM 3: out_seq = ynorm @ W_out -> bf16 [NTOK, 256]
__global__ void __launch_bounds__(256)
k_gemm_out(const __bf16* __restrict__ A, const __bf16* __restrict__ B,
           __bf16* __restrict__ oseq)
{
    __shared__ __align__(16) __bf16 sm[(GBLK_M + GBLK_N) * LDSS];
    const int blockM = blockIdx.y * GBLK_M, blockN = blockIdx.x * GBLK_N;
    v8f acc0 = {}, acc1 = {};
    gemm_tile<D_INNER, DM>(A, B, blockM, blockN, sm, sm + GBLK_M * LDSS, acc0, acc1);
    GEMM_EPILOG_COORDS();
#pragma unroll
    for (int r = 0; r < 8; ++r) {
        oseq[(size_t)(gm0 + r) * DM + gn0] = (__bf16)acc0[r];
        oseq[(size_t)(gm0 + r) * DM + gn1] = (__bf16)acc1[r];
    }
}

// GEMM 4: unembed + bias + residual, scattered to (B,23,T,128,128) fp32
__global__ void __launch_bounds__(256)
k_gemm_unembed(const __bf16* __restrict__ A, const __bf16* __restrict__ B,
               const float* __restrict__ ub, const float* __restrict__ xin,
               float* __restrict__ out)
{
    __shared__ __align__(16) __bf16 sm[(GBLK_M + GBLK_N) * LDSS];
    const int blockM = blockIdx.y * GBLK_M, blockN = blockIdx.x * GBLK_N;
    v8f acc0 = {}, acc1 = {};
    gemm_tile<DM, NUN_PAD>(A, B, blockM, blockN, sm, sm + GBLK_M * LDSS, acc0, acc1);
    GEMM_EPILOG_COORDS();
#pragma unroll
    for (int half = 0; half < 2; ++half) {
        const int gn = half ? gn1 : gn0;
        if (gn >= KPATCH) continue;
        int c = gn >> 4, q = gn & 15, i = q >> 2, j = q & 3;
        float bv = ub[c];
#pragma unroll
        for (int r = 0; r < 8; ++r) {
            int gm = gm0 + r;
            int b = gm >> 13, rr = gm & 8191;
            int t = rr >> 10, r2 = rr & 1023, hp = r2 >> 5, wp = r2 & 31;
            size_t addr = ((((size_t)(b * 23 + c) * T_SZ + t) * 128 + hp * 4 + i) * 128)
                          + wp * 4 + j;
            float a = half ? acc1[r] : acc0[r];
            out[addr] = a + bv + xin[addr];
        }
    }
}

// ---------------------------------------------------------------------
// causal depthwise conv (D_CONV=4) over L per batch + SiLU. block=544 (=17 waves)
__global__ void k_conv(const float* __restrict__ zx, const float* __restrict__ cw,
                       const float* __restrict__ cb, float* __restrict__ xbc)
{
    int m  = blockIdx.x;            // token
    int ch = threadIdx.x;           // 0..543
    int b = m >> 13, l = m & 8191;
    float acc = cb[ch];
#pragma unroll
    for (int k = 0; k < 4; ++k) {
        int ls = l + k - 3;
        if (ls >= 0)
            acc += zx[((size_t)b * L_SEQ + ls) * D_IN_PROJ + D_INNER + ch] * cw[ch * 4 + k];
    }
    xbc[(size_t)m * CONV_DIM + ch] = acc / (1.f + expf(-acc));   // SiLU
}

// dt = softplus(dt + dt_bias);  dA = exp(dt * -exp(A_log))
__global__ void k_dtprep(const float* __restrict__ zx, const float* __restrict__ dtb,
                         const float* __restrict__ alog,
                         float* __restrict__ dto, float* __restrict__ dao)
{
    int idx = blockIdx.x * blockDim.x + threadIdx.x;
    if (idx >= NTOK * NHEADS) return;
    int h = idx & 7;
    size_t m = (size_t)idx >> 3;
    float d  = zx[m * D_IN_PROJ + (D_INNER + CONV_DIM) + h] + dtb[h];
    float sp = (d > 20.f) ? d : log1pf(expf(d));
    float Ah = -expf(alog[h]);
    dto[idx] = sp;
    dao[idx] = expf(sp * Ah);
}

// sequential SSM scan: 32 blocks = (b,head); state h[64x16] in registers.
// thread t: p = t/4, owns 4 states n = (t%4)*4.. ; quad shuffle-reduce for y.
// Register double-buffering: iteration l+1 loads are issued before computing l.
__global__ void __launch_bounds__(256)
k_scan(const float* __restrict__ xbc, const float* __restrict__ dtb,
       const float* __restrict__ dab, const float* __restrict__ Dp_,
       float* __restrict__ yb)
{
    int blk = blockIdx.x;
    int b = blk >> 3, h = blk & 7;
    int tid = threadIdx.x;
    int p  = tid >> 2;
    int ng = (tid & 3) * 4;
    float h0 = 0.f, h1 = 0.f, h2 = 0.f, h3 = 0.f;
    float Dp = Dp_[h];
    size_t m = (size_t)b * L_SEQ;

    // preload iteration 0
    const float* row = xbc + m * CONV_DIM;
    float dA  = dab[m * NHEADS + h];
    float dtv = dtb[m * NHEADS + h];
    float xv  = row[h * HEADDIM + p];
    float B0 = row[D_INNER + ng + 0], B1 = row[D_INNER + ng + 1];
    float B2 = row[D_INNER + ng + 2], B3 = row[D_INNER + ng + 3];
    float C0 = row[D_INNER + D_STATE + ng + 0], C1 = row[D_INNER + D_STATE + ng + 1];
    float C2 = row[D_INNER + D_STATE + ng + 2], C3 = row[D_INNER + D_STATE + ng + 3];

    for (int l = 0; l < L_SEQ; ++l) {
        // prefetch several rows ahead into cache (global_prefetch_b8)
        if (l + 8 < L_SEQ) {
            const float* frow = xbc + (m + 8) * CONV_DIM;
            __builtin_prefetch(frow + h * HEADDIM + p, 0, 0);
            __builtin_prefetch(frow + D_INNER + ng, 0, 0);
        }
        // issue next iteration's loads
        float ndA = 0.f, ndt = 0.f, nxv = 0.f;
        float nB0 = 0.f, nB1 = 0.f, nB2 = 0.f, nB3 = 0.f;
        float nC0 = 0.f, nC1 = 0.f, nC2 = 0.f, nC3 = 0.f;
        if (l + 1 < L_SEQ) {
            const float* nrow = xbc + (m + 1) * CONV_DIM;
            ndA = dab[(m + 1) * NHEADS + h];
            ndt = dtb[(m + 1) * NHEADS + h];
            nxv = nrow[h * HEADDIM + p];
            nB0 = nrow[D_INNER + ng + 0]; nB1 = nrow[D_INNER + ng + 1];
            nB2 = nrow[D_INNER + ng + 2]; nB3 = nrow[D_INNER + ng + 3];
            nC0 = nrow[D_INNER + D_STATE + ng + 0]; nC1 = nrow[D_INNER + D_STATE + ng + 1];
            nC2 = nrow[D_INNER + D_STATE + ng + 2]; nC3 = nrow[D_INNER + D_STATE + ng + 3];
        }
        // recurrence
        float xdt = xv * dtv;
        h0 = fmaf(dA, h0, xdt * B0);
        h1 = fmaf(dA, h1, xdt * B1);
        h2 = fmaf(dA, h2, xdt * B2);
        h3 = fmaf(dA, h3, xdt * B3);
        float part = h0 * C0 + h1 * C1 + h2 * C2 + h3 * C3;
        part += __shfl_xor(part, 1, 32);
        part += __shfl_xor(part, 2, 32);
        if ((tid & 3) == 0)
            yb[m * D_INNER + h * HEADDIM + p] = part + Dp * xv;
        // rotate buffers
        dA = ndA; dtv = ndt; xv = nxv;
        B0 = nB0; B1 = nB1; B2 = nB2; B3 = nB3;
        C0 = nC0; C1 = nC1; C2 = nC2; C3 = nC3;
        ++m;
    }
}

// gate (y * silu(z)) + RMSNorm(512) * norm_w -> bf16. one block per token.
__global__ void __launch_bounds__(256)
k_gate_norm(const float* __restrict__ yb, const float* __restrict__ zx,
            const float* __restrict__ nw, __bf16* __restrict__ yn)
{
    size_t m = blockIdx.x;
    int tid = threadIdx.x;
    __shared__ float red[8];
    int d0 = tid, d1 = tid + 256;
    float z0 = zx[m * D_IN_PROJ + d0];
    float z1 = zx[m * D_IN_PROJ + d1];
    float v0 = yb[m * D_INNER + d0] * (z0 / (1.f + expf(-z0)));
    float v1 = yb[m * D_INNER + d1] * (z1 / (1.f + expf(-z1)));
    float ss = v0 * v0 + v1 * v1;
#pragma unroll
    for (int off = 16; off > 0; off >>= 1) ss += __shfl_xor(ss, off, 32);
    if ((tid & 31) == 0) red[tid >> 5] = ss;
    __syncthreads();
    float tot = red[0] + red[1] + red[2] + red[3] + red[4] + red[5] + red[6] + red[7];
    float rs = rsqrtf(tot * (1.f / 512.f) + 1e-5f);
    yn[m * D_INNER + d0] = (__bf16)(v0 * rs * nw[d0]);
    yn[m * D_INNER + d1] = (__bf16)(v1 * rs * nw[d1]);
}

// =====================================================================
extern "C" void kernel_launch(void* const* d_in, const int* in_sizes, int n_in,
                              void* d_out, int out_size, void* d_ws, size_t ws_size,
                              hipStream_t stream)
{
    (void)in_sizes; (void)n_in; (void)out_size; (void)ws_size;
    const float* x         = (const float*)d_in[0];
    const float* patch_w   = (const float*)d_in[1];
    const float* patch_b   = (const float*)d_in[2];
    const float* W_in      = (const float*)d_in[3];
    const float* conv_w    = (const float*)d_in[4];
    const float* conv_b    = (const float*)d_in[5];
    const float* dt_bias   = (const float*)d_in[6];
    const float* A_log     = (const float*)d_in[7];
    const float* D_param   = (const float*)d_in[8];
    const float* norm_w    = (const float*)d_in[9];
    const float* W_out     = (const float*)d_in[10];
    const float* unembed_w = (const float*)d_in[11];
    const float* unembed_b = (const float*)d_in[12];
    float* out = (float*)d_out;

    char* ws = (char*)d_ws;
    __bf16* Ap    = (__bf16*)(ws + OFF_APATCH);
    __bf16* Bp    = (__bf16*)(ws + OFF_BPATCH);
    __bf16* xs    = (__bf16*)(ws + OFF_XS);
    __bf16* Winb  = (__bf16*)(ws + OFF_WINB);
    float*  zx    = (float* )(ws + OFF_ZX);
    float*  xbc   = (float* )(ws + OFF_XBC);
    float*  dtbuf = (float* )(ws + OFF_DT);
    float*  dabuf = (float* )(ws + OFF_DA);
    float*  ybuf  = (float* )(ws + OFF_YB);
    __bf16* ynorm = (__bf16*)(ws + OFF_YN);
    __bf16* Woutb = (__bf16*)(ws + OFF_WOUTB);
    __bf16* oseq  = (__bf16*)(ws + OFF_OSEQ);
    __bf16* Bun   = (__bf16*)(ws + OFF_BUN);

    // 1. patch embed (im2col + WMMA GEMM + bias)
    k_im2col<<<(unsigned)(((size_t)NTOK * KPATCH_PAD + 255) / 256), 256, 0, stream>>>(x, Ap);
    k_prep_bpatch<<<(KPATCH_PAD * DM + 255) / 256, 256, 0, stream>>>(patch_w, Bp);
    k_gemm_patch<<<dim3(DM / GBLK_N, NTOK / GBLK_M), 256, 0, stream>>>(Ap, Bp, patch_b, xs);

    // 2. in_proj
    k_prep_win<<<(DM * NPROJ_PAD + 255) / 256, 256, 0, stream>>>(W_in, Winb);
    k_gemm_inproj<<<dim3(NPROJ_PAD / GBLK_N, NTOK / GBLK_M), 256, 0, stream>>>(xs, Winb, zx);

    // 3. conv + SiLU, dt/dA
    k_conv<<<NTOK, CONV_DIM, 0, stream>>>(zx, conv_w, conv_b, xbc);
    k_dtprep<<<(NTOK * NHEADS + 255) / 256, 256, 0, stream>>>(zx, dt_bias, A_log, dtbuf, dabuf);

    // 4. sequential SSM scan
    k_scan<<<B_SZ * NHEADS, 256, 0, stream>>>(xbc, dtbuf, dabuf, D_param, ybuf);

    // 5. gate + RMSNorm
    k_gate_norm<<<NTOK, 256, 0, stream>>>(ybuf, zx, norm_w, ynorm);

    // 6. out projection
    k_prep_wout<<<(D_INNER * DM + 255) / 256, 256, 0, stream>>>(W_out, Woutb);
    k_gemm_out<<<dim3(DM / GBLK_N, NTOK / GBLK_M), 256, 0, stream>>>(ynorm, Woutb, oseq);

    // 7. unembed + bias + residual (fused scatter)
    k_prep_bun<<<(DM * NUN_PAD + 255) / 256, 256, 0, stream>>>(unembed_w, Bun);
    k_gemm_unembed<<<dim3(NUN_PAD / GBLK_N, NTOK / GBLK_M), 256, 0, stream>>>(oseq, Bun,
                                                                unembed_b, x, out);
}
